// NonLocalModel_66288525246798
// MI455X (gfx1250) — matrix operand order repaired and checked by reading, not compiled, not run on hardware
//
#include <hip/hip_runtime.h>
#include <math.h>

// ---------------------------------------------------------------------------
// Problem constants (match reference)
// ---------------------------------------------------------------------------
#define NN   2048      // fine grid points
#define MM   128       // coarse points
#define INF_ 256       // input features
#define HH   128       // hidden
#define K9   9         // 1 + 3 + 5 sh components
#define KTOT (MM * K9) // 1152 : contraction length of final GEMM

#define A2B_     1.88972999f
#define MIN_STD_ (0.32f * A2B_ * 0.5f)
#define MAX_STD_ (2.32f * A2B_ * 0.5f)

typedef _Float16 h16;
typedef __attribute__((ext_vector_type(16))) _Float16 v16h;
typedef __attribute__((ext_vector_type(8)))  _Float16 v8h;
typedef __attribute__((ext_vector_type(8)))  float    v8f;

union H16x16 { v16h v; v8h h[2]; };

// ---------------------------------------------------------------------------
// Async global->LDS staging (CDNA5 ASYNCcnt path) with sync fallback
// ---------------------------------------------------------------------------
#if __has_builtin(__builtin_amdgcn_global_load_async_to_lds_b128)
#define ASYNC_LDS 1
#else
#define ASYNC_LDS 0
#endif

typedef int v4i_ __attribute__((vector_size(16)));
typedef __attribute__((address_space(1))) v4i_* gptr128;
typedef __attribute__((address_space(3))) v4i_* lptr128;

__device__ __forceinline__ void stage16(const h16* g, h16* l) {
#if ASYNC_LDS
  __builtin_amdgcn_global_load_async_to_lds_b128(
      (gptr128)const_cast<h16*>(g), (lptr128)l, 0, 0);
#else
  *(v8h*)l = *(const v8h*)g;           // ds_store_b128 fallback
#endif
}

__device__ __forceinline__ void stage_wait() {
#if ASYNC_LDS
#if __has_builtin(__builtin_amdgcn_s_wait_asynccnt)
  __builtin_amdgcn_s_wait_asynccnt(0);
#else
  asm volatile("s_wait_asynccnt 0x0" ::: "memory");
#endif
#endif
}

// ---------------------------------------------------------------------------
// WMMA fragment loaders (gfx1250 wave32, 16x16x32 f16)
//   A (16x32, MxK), row-major A[., lda]:
//     lane l: row = l&15, kb = (l>=16)?8:0 ; halves 0..7 = K kb..kb+7,
//     halves 8..15 = K 16+kb..16+kb+7  -> two contiguous 16B loads
//   B (32x16, KxN) consumed from BT row-major [N, ldb]:
//     lane l: col = l&15, kb = (l>=16)?8:0 ; halves 0..7 = K kb..kb+7, halves 8..15 = K 16+kb..16+kb+7
// ---------------------------------------------------------------------------
__device__ __forceinline__ v16h ld_a_frag(const h16* A, int lda, int lane) {
  const int r  = lane & 15;
  const int kb = (lane >> 4) << 3;
  const h16* p = A + r * lda + kb;
  H16x16 t;
  t.h[0] = *(const v8h*)(p);
  t.h[1] = *(const v8h*)(p + 16);
  return t.v;
}

__device__ __forceinline__ v16h ld_bt_frag_lds(const h16* p) {
  H16x16 t;
  t.h[0] = *(const v8h*)(p);
  t.h[1] = *(const v8h*)(p + 16);
  return t.v;
}

__device__ __forceinline__ float silu_f(float x) {
  return x / (1.0f + __expf(-x));
}

// ---------------------------------------------------------------------------
// Blocked WMMA GEMM: block = 8 waves = 128 rows x 128 cols output macro-tile.
// B slab (128 cols x 32 K, 8KB f16) staged to LDS per k-step (async),
// reused by all 8 waves; each wave reuses one A fragment across 8 WMMAs.
//   mode 0: out f16 = silu(acc + bias)          (pre-linear)
//   mode 1: out f32 = acc * gw[row]             (down projection)
//   mode 2: out f32 = silu(acc + bias)          (final GEMM)
// ---------------------------------------------------------------------------
__global__ void nlm_gemm128(const h16* __restrict__ A, const h16* __restrict__ BT,
                            const float* __restrict__ bias, const float* __restrict__ gw,
                            void* __restrict__ outp, int NcTot, int K, int mode) {
  __shared__ h16 lds_b[128 * 32];
  const int tid   = threadIdx.x;
  const int lane  = tid & 31;
  const int wv    = tid >> 5;                    // 0..7
  const int rbase = blockIdx.x * 128 + wv * 16;
  const int cb    = blockIdx.y * 128;
  const h16* Arow = A + (size_t)rbase * K;
  const h16* Bblk = BT + (size_t)cb * K;

  v8f zero = {};
  v8f acc[8];
  for (int t = 0; t < 8; ++t) acc[t] = zero;

  const int c0 = lane & 15;
  const int kb = (lane >> 4) << 3;

  for (int k = 0; k < K; k += 32) {
    __syncthreads();                             // previous slab fully consumed
    // stage B slab: 512 chunks of 16B; 2 per thread
    for (int r = 0; r < 2; ++r) {
      const int ch   = tid + r * 256;
      const int col  = ch >> 2;
      const int part = ch & 3;
      stage16(Bblk + col * K + k + part * 8, &lds_b[col * 32 + part * 8]);
    }
    stage_wait();
    __syncthreads();                             // slab published

    v16h a = ld_a_frag(Arow + k, K, lane);
    for (int ct = 0; ct < 8; ++ct) {
      v16h b = ld_bt_frag_lds(&lds_b[(ct * 16 + c0) * 32 + kb]);
      acc[ct] = __builtin_amdgcn_wmma_f32_16x16x32_f16(
          false, a, false, b, (short)0, acc[ct], false, false);
    }
  }

  const int r0 = (lane >> 4) << 3;
  for (int ct = 0; ct < 8; ++ct) {
    const int col = cb + ct * 16 + c0;
    for (int i = 0; i < 8; ++i) {
      const int row = rbase + r0 + i;
      const float v = acc[ct][i];
      if (mode == 0) {
        ((h16*)outp)[(size_t)row * NcTot + col] = (h16)silu_f(v + bias[col]);
      } else if (mode == 1) {
        ((float*)outp)[(size_t)row * NcTot + col] = v * gw[row];
      } else {
        ((float*)outp)[(size_t)row * NcTot + col] = silu_f(v + bias[col]);
      }
    }
  }
}

// ---------------------------------------------------------------------------
// Converters
// ---------------------------------------------------------------------------
__global__ void nlm_cvt_f16(const float* __restrict__ in, h16* __restrict__ out, int n) {
  int i = blockIdx.x * blockDim.x + threadIdx.x;
  if (i < n) out[i] = (h16)in[i];
}

// out[c*R + r] = in[r*C + c]   (transpose + f16 convert)
__global__ void nlm_tr_f16(const float* __restrict__ in, h16* __restrict__ out, int R, int C) {
  int i = blockIdx.x * blockDim.x + threadIdx.x;
  if (i >= R * C) return;
  int r = i / C, c = i - r * C;
  out[c * R + r] = (h16)in[r * C + c];
}

// ---------------------------------------------------------------------------
// Geometry: norm-scaled spherical harmonics (sh_l / sqrt(2l+1)) and d^2
// ---------------------------------------------------------------------------
__global__ void nlm_geom(const float* __restrict__ gc, const float* __restrict__ cc,
                         h16* __restrict__ shs, float* __restrict__ d2out) {
  int idx = blockIdx.x * blockDim.x + threadIdx.x;
  if (idx >= NN * MM) return;
  int n = idx >> 7, m = idx & (MM - 1);
  float dx = gc[n * 3 + 0] - cc[m * 3 + 0];
  float dy = gc[n * 3 + 1] - cc[m * 3 + 1];
  float dz = gc[n * 3 + 2] - cc[m * 3 + 2];
  float d2 = dx * dx + dy * dy + dz * dz + 1e-20f;
  float ri = rsqrtf(d2);
  float x = dx * ri, y = dy * ri, z = dz * ri;
  d2out[idx] = d2;
  const float IS3 = 0.57735026919f;     // 1/sqrt(3)
  const float C35 = 0.77459666924f;     // sqrt(3/5)
  const float IS5 = 0.44721359550f;     // 1/sqrt(5)
  h16* s = shs + n * KTOT + m * K9;
  s[0] = (h16)1.0f;
  s[1] = (h16)(x * IS3);
  s[2] = (h16)(y * IS3);
  s[3] = (h16)(z * IS3);
  s[4] = (h16)(C35 * x * z);
  s[5] = (h16)(C35 * x * y);
  s[6] = (h16)((y * y - 0.5f * (x * x + z * z)) * IS5);
  s[7] = (h16)(C35 * y * z);
  s[8] = (h16)(0.5f * C35 * (z * z - x * x));
}

// ---------------------------------------------------------------------------
// Down contraction (radial gating computed on the fly, never materialized):
//   c[(m*9+j), u] = sum_n hwg[n, l(j)*128+u] * rad(d2[n,m], u) * shs[n,m,j]
//   rad = C1[u] * d2 * exp(-d2 * C2[u])
// ---------------------------------------------------------------------------
__global__ void nlm_down(const float* __restrict__ hwg, const float* __restrict__ d2arr,
                         const h16* __restrict__ shs, float* __restrict__ c_out) {
  const int m = blockIdx.x;
  const int u = threadIdx.x;
  const float s = MIN_STD_ + (MAX_STD_ - MIN_STD_) * (float)u / (float)(HH - 1);
  const float temps = 2.0f * s * s;
  const float C2 = 1.0f / temps;
  const float C1 = (2.0f / 3.0f) * C2 * __powf(3.14159265358979f * temps, -1.5f);
  float acc[K9];
  for (int j = 0; j < K9; ++j) acc[j] = 0.0f;
  for (int n = 0; n < NN; ++n) {
    const float d2  = d2arr[n * MM + m];                  // uniform -> scalar load
    const float rad = C1 * d2 * __expf(-d2 * C2);
    const float* hb = hwg + n * (3 * HH);
    const float a0 = hb[u]          * rad;
    const float a1 = hb[HH + u]     * rad;
    const float a2 = hb[2 * HH + u] * rad;
    const h16* s9 = shs + n * KTOT + m * K9;              // uniform -> scalar loads
    acc[0] += a0 * (float)s9[0];
    acc[1] += a1 * (float)s9[1];
    acc[2] += a1 * (float)s9[2];
    acc[3] += a1 * (float)s9[3];
    acc[4] += a2 * (float)s9[4];
    acc[5] += a2 * (float)s9[5];
    acc[6] += a2 * (float)s9[6];
    acc[7] += a2 * (float)s9[7];
    acc[8] += a2 * (float)s9[8];
  }
  for (int j = 0; j < K9; ++j) c_out[(m * K9 + j) * HH + u] = acc[j];
}

// ---------------------------------------------------------------------------
// Fold W_up (per-l) and W_post into cuW^T (f16, [HH, KTOT] = B^T of final GEMM)
// ---------------------------------------------------------------------------
__global__ void nlm_fold_up(const float* __restrict__ c_f32, const float* __restrict__ W_up,
                            const float* __restrict__ W_post, h16* __restrict__ cuWT) {
  const int m = blockIdx.x;   // 128
  const int w = threadIdx.x;  // 128
  __shared__ float cu[K9][HH];
  for (int j = 0; j < K9; ++j) {
    const int l = (j == 0) ? 0 : ((j < 4) ? 1 : 2);
    const float* cr = c_f32 + (m * K9 + j) * HH;
    const float* Wl = W_up + l * HH * HH;
    float a = 0.0f;
    for (int u = 0; u < HH; ++u) a += cr[u] * Wl[u * HH + w];
    cu[j][w] = a;
  }
  __syncthreads();
  for (int j = 0; j < K9; ++j) {
    float a = 0.0f;
    for (int u = 0; u < HH; ++u) a += cu[j][u] * W_post[u * HH + w];
    cuWT[w * KTOT + m * K9 + j] = (h16)a;
  }
}

// ---------------------------------------------------------------------------
// Launch
// ---------------------------------------------------------------------------
extern "C" void kernel_launch(void* const* d_in, const int* in_sizes, int n_in,
                              void* d_out, int out_size, void* d_ws, size_t ws_size,
                              hipStream_t stream) {
  const float* h      = (const float*)d_in[0];
  const float* gc     = (const float*)d_in[1];
  const float* cc     = (const float*)d_in[2];
  const float* gw     = (const float*)d_in[3];
  const float* W_pre  = (const float*)d_in[4];
  const float* b_pre  = (const float*)d_in[5];
  const float* W_down = (const float*)d_in[6];
  const float* W_up   = (const float*)d_in[7];
  const float* W_post = (const float*)d_in[8];
  const float* b_post = (const float*)d_in[9];
  float* out = (float*)d_out;

  char* ws = (char*)d_ws;
  size_t off = 0;
  auto alloc = [&](size_t bytes) -> void* {
    void* p = ws + off;
    off += (bytes + 255) & ~(size_t)255;
    return p;
  };
  h16*   h_h    = (h16*)  alloc((size_t)NN * INF_ * 2);   // h in f16
  h16*   WpreT  = (h16*)  alloc((size_t)HH * INF_ * 2);   // W_pre^T f16 [HH,INF]
  h16*   WdT    = (h16*)  alloc((size_t)3 * HH * HH * 2); // W_down^T f16 [3*HH, HH]
  h16*   hpre   = (h16*)  alloc((size_t)NN * HH * 2);     // silu(h@W_pre+b) f16
  float* hwg    = (float*)alloc((size_t)NN * 3 * HH * 4); // (hpre@W_down[l])*gw
  h16*   shs    = (h16*)  alloc((size_t)NN * KTOT * 2);   // scaled sph harmonics
  float* d2     = (float*)alloc((size_t)NN * MM * 4);     // squared distances
  float* c_f32  = (float*)alloc((size_t)KTOT * HH * 4);   // coarse coefficients
  h16*   cuWT   = (h16*)  alloc((size_t)HH * KTOT * 2);   // folded up*post weights
  (void)ws_size; (void)in_sizes; (void)n_in; (void)out_size;

  // Stage 0: conversions / transposes
  nlm_cvt_f16<<<(NN * INF_ + 255) / 256, 256, 0, stream>>>(h, h_h, NN * INF_);
  nlm_tr_f16<<<(INF_ * HH + 255) / 256, 256, 0, stream>>>(W_pre, WpreT, INF_, HH);
  for (int l = 0; l < 3; ++l)
    nlm_tr_f16<<<(HH * HH + 255) / 256, 256, 0, stream>>>(
        W_down + l * HH * HH, WdT + l * HH * HH, HH, HH);

  // Stage 1: geometry
  nlm_geom<<<(NN * MM + 255) / 256, 256, 0, stream>>>(gc, cc, shs, d2);

  // Stage 2: WMMA pre-GEMM + silu  (2048x128x256)
  nlm_gemm128<<<dim3(NN / 128, 1), 256, 0, stream>>>(
      h_h, WpreT, b_pre, nullptr, hpre, HH, INF_, 0);

  // Stage 3: WMMA down projection * gw  (2048x384x128); blockIdx.y = l
  nlm_gemm128<<<dim3(NN / 128, 3), 256, 0, stream>>>(
      hpre, WdT, nullptr, gw, hwg, 3 * HH, HH, 1);

  // Stage 4: radial-gated contraction onto coarse nodes (exp on the fly)
  nlm_down<<<MM, HH, 0, stream>>>(hwg, d2, shs, c_f32);

  // Stage 5: fold W_up and W_post into cuW^T
  nlm_fold_up<<<MM, HH, 0, stream>>>(c_f32, W_up, W_post, cuWT);

  // Stage 6: WMMA final GEMM (2048x128, K=1152) + bias + silu
  nlm_gemm128<<<dim3(NN / 128, 1), 256, 0, stream>>>(
      shs, cuWT, b_post, nullptr, out, HH, KTOT, 2);
}
